// SpikeRegressionNet_71073118814296
// MI455X (gfx1250) — matrix operand order repaired and checked by reading, not compiled, not run on hardware
//
#include <hip/hip_runtime.h>

typedef __attribute__((ext_vector_type(16))) _Float16     v16h;
typedef __attribute__((ext_vector_type(8)))  _Float16     v8h;
typedef __attribute__((ext_vector_type(8)))  float        v8f;
typedef __attribute__((ext_vector_type(4)))  float        v4f;
typedef __attribute__((ext_vector_type(4)))  unsigned int v4u;

#define T_STEPS 64
#define BATCH   4096
#define HID     256
#define OUTD    2
#define MTILE   32                 // batch rows per workgroup
#define NBLK    (BATCH / MTILE)    // 128 workgroups

// Shared memory layout (bytes):
//  WhL   : 256*256 f16 = 131072
//  Wh2L  : 256*256 f16 = 131072
//  sA1   :  32*256 f16 =  16384
//  sA2   :  32*256 f16 =  16384
//  woL   :   2*256 f32 =   2048
//  wpL   :   2*256 f32 =   2048
//  pL    :   9*256 f32 =   9216   (beta/thr/bias for 3 layers, beta pre-clamped)
//  wiL   :   256*3 f32 =   3072
//  xL    :    32*3 f32 =    384
//  recL  :    32*4 f32 =    512
#define SMEM_BYTES (131072 + 131072 + 16384 + 16384 + 2048 + 2048 + 9216 + 3072 + 384 + 512)

struct Frag32 { v4u lo, hi; };     // 32B = one v16h worth of data, loaded as 2x b128

__global__ __launch_bounds__(256) void snn_scan_kernel(
    const float* __restrict__ x,
    const float* __restrict__ W_in,  const float* __restrict__ b_in,
    const float* __restrict__ beta_in, const float* __restrict__ thr_in,
    const float* __restrict__ W_h,   const float* __restrict__ b_h,
    const float* __restrict__ beta_h,  const float* __restrict__ thr_h,
    const float* __restrict__ W_h2,  const float* __restrict__ b_h2,
    const float* __restrict__ beta_h2, const float* __restrict__ thr_h2,
    const float* __restrict__ W_out, const float* __restrict__ b_out,
    const float* __restrict__ beta_out,
    const float* __restrict__ W_pred, const float* __restrict__ b_pred,
    float* __restrict__ out)
{
    extern __shared__ char smem[];
    _Float16* WhL  = (_Float16*)smem;            // [HID][HID]  row = out feature n, col = k
    _Float16* Wh2L = WhL  + HID * HID;
    _Float16* sA1  = Wh2L + HID * HID;           // [MTILE][HID] spike staging (s1, then s4)
    _Float16* sA2  = sA1  + MTILE * HID;         // [MTILE][HID] spike staging (s2)
    float* woL  = (float*)(sA2 + MTILE * HID);   // [OUTD][HID]
    float* wpL  = woL + OUTD * HID;              // [OUTD][T*4]
    float* pL   = wpL + OUTD * HID;              // [9][HID]
    float* wiL  = pL  + 9 * HID;                 // [HID][3]
    float* xL   = wiL + HID * 3;                 // [MTILE][3]
    float* recL = xL + MTILE * 3;                // [MTILE][4]

    const int tid    = threadIdx.x;
    const int lane   = tid & 31;
    const int lane16 = lane & 15;
    const int laneHi = lane >> 4;                // 0 or 1
    const int wave   = tid >> 5;                 // 0..7
    const int waveM  = wave & 1;                 // which 16-row half of the 32-row tile
    const int nBase  = (wave >> 1) * 64;         // 64 output columns per wave (4 x 16)
    const int wgB    = blockIdx.x * MTILE;       // global batch base

    // ---- stage recurrent weights (fp32 -> fp16) into LDS ----
    for (int i = tid; i < HID * HID; i += 256) {
        WhL[i]  = (_Float16)W_h[i];
        Wh2L[i] = (_Float16)W_h2[i];
    }
    for (int i = tid; i < OUTD * HID; i += 256) {
        woL[i] = W_out[i];
        wpL[i] = W_pred[i];
    }
    // ---- stage per-neuron LIF parameters (keeps them out of persistent VGPRs) ----
    {
        int n = tid;                              // exactly 256 threads, one neuron each
        pL[0 * HID + n] = fminf(fmaxf(beta_in[n], 0.f), 1.f);
        pL[1 * HID + n] = thr_in[n];
        pL[2 * HID + n] = b_in[n];
        pL[3 * HID + n] = fminf(fmaxf(beta_h[n], 0.f), 1.f);
        pL[4 * HID + n] = thr_h[n];
        pL[5 * HID + n] = b_h[n];
        pL[6 * HID + n] = fminf(fmaxf(beta_h2[n], 0.f), 1.f);
        pL[7 * HID + n] = thr_h2[n];
        pL[8 * HID + n] = b_h2[n];
        wiL[n * 3 + 0]  = W_in[n * 3 + 0];
        wiL[n * 3 + 1]  = W_in[n * 3 + 1];
        wiL[n * 3 + 2]  = W_in[n * 3 + 2];
    }

    // ---- persistent LIF state lives in WMMA C-fragment registers ----
    v8f m1[4], m2[4], m4[4];
    const v8f vzero = {0.f, 0.f, 0.f, 0.f, 0.f, 0.f, 0.f, 0.f};
#pragma unroll
    for (int ti = 0; ti < 4; ++ti) { m1[ti] = vzero; m2[ti] = vzero; m4[ti] = vzero; }

    // output-layer: thread tid handles (b = tid>>3, o = (tid>>2)&1, kchunk = tid&3);
    // the kchunk==0 thread of each (b,o) group owns m3/acc state.
    const int ob = tid >> 3, oo = (tid >> 2) & 1, kc = tid & 3;
    float m3 = 0.f, accP = 0.f, betaO = 0.f, biasO = 0.f;
    if (kc == 0) {
        betaO = fminf(fmaxf(beta_out[oo], 0.f), 1.f);
        biasO = b_out[oo];
    }

    // Gaussian temporal embedding constants: center=32, sigma=6.4
    const float sigma = (float)T_STEPS / 10.0f;
    const float e0    = -0.5f * (32.f / sigma) * (32.f / sigma);
    const float emin  = __expf(e0);
    const float einv  = 1.f / (1.f - emin);

    __syncthreads();

    const unsigned int* sA1u = (const unsigned int*)sA1;
    const unsigned int* sA2u = (const unsigned int*)sA2;
    const unsigned int* WhU  = (const unsigned int*)WhL;
    const unsigned int* Wh2U = (const unsigned int*)Wh2L;
    const int rowBase = waveM * 16;
    const int arow    = rowBase + lane16;        // A-fragment row for this lane

    for (int t = 0; t < T_STEPS; ++t) {
        float dt   = ((float)t - 32.f) / sigma;
        float embt = (__expf(-0.5f * dt * dt) - emin) * einv;

        // stage weighted x[t] tile: 32 rows x 3 features
        if (tid < MTILE * 3) {
            int bl = tid / 3, k = tid - bl * 3;
            xL[tid] = x[(size_t)t * BATCH * 3 + (size_t)(wgB + bl) * 3 + k] * embt;
        }
        __syncthreads();                         // B1: xL ready; prior recL/sA1 reads done

        // ---- layer 1: tiny input GEMM (K=3) + LIF, spikes -> sA1 ----
#pragma unroll
        for (int ti = 0; ti < 4; ++ti) {
            int nn = nBase + ti * 16 + lane16;
            float bt = pL[0 * HID + nn], th = pL[1 * HID + nn], bb = pL[2 * HID + nn];
            float w0 = wiL[nn * 3 + 0], w1 = wiL[nn * 3 + 1], w2 = wiL[nn * 3 + 2];
#pragma unroll
            for (int r = 0; r < 8; ++r) {
                int bloc = rowBase + r + laneHi * 8;
                float cur = bb + xL[bloc * 3 + 0] * w0
                               + xL[bloc * 3 + 1] * w1
                               + xL[bloc * 3 + 2] * w2;
                float m = m1[ti][r];
                float rst = (m > th) ? th : 0.f;     // subtract-reset from prev mem
                m = bt * m + cur - rst;
                m1[ti][r] = m;
                sA1[bloc * HID + nn] = (_Float16)((m > th) ? 1.f : 0.f);
            }
        }
        __syncthreads();                         // B2: s1 staged

        // ---- layer 2: s1 @ W_h^T via WMMA f16 (K=256 in 8 steps of 32) ----
        v8f acc[4];
#pragma unroll
        for (int ti = 0; ti < 4; ++ti) acc[ti] = vzero;
#pragma unroll
        for (int kk = 0; kk < 8; ++kk) {
            Frag32 af;                           // A dwords {0..3, 8..11} + laneHi*4 -> 2x b128
            af.lo = *(const v4u*)(sA1u + arow * 128 + kk * 16 + laneHi * 4);
            af.hi = *(const v4u*)(sA1u + arow * 128 + kk * 16 + laneHi * 4 + 8);
            v16h a = __builtin_bit_cast(v16h, af);
#pragma unroll
            for (int ti = 0; ti < 4; ++ti) {
                int nn = nBase + ti * 16 + lane16;
                Frag32 bf;                       // B dwords {0..3, 8..11} + laneHi*4 -> 2x b128
                bf.lo = *(const v4u*)(WhU + nn * 128 + kk * 16 + laneHi * 4);
                bf.hi = *(const v4u*)(WhU + nn * 128 + kk * 16 + laneHi * 4 + 8);
                v16h b = __builtin_bit_cast(v16h, bf);
                acc[ti] = __builtin_amdgcn_wmma_f32_16x16x32_f16(
                    false, a, false, b, (short)0, acc[ti], false, false);
            }
        }
        // LIF2, spikes -> sA2
#pragma unroll
        for (int ti = 0; ti < 4; ++ti) {
            int nn = nBase + ti * 16 + lane16;
            float bt = pL[3 * HID + nn], th = pL[4 * HID + nn], bb = pL[5 * HID + nn];
#pragma unroll
            for (int r = 0; r < 8; ++r) {
                int bloc = rowBase + r + laneHi * 8;
                float cur = acc[ti][r] + bb;
                float m = m2[ti][r];
                float rst = (m > th) ? th : 0.f;
                m = bt * m + cur - rst;
                m2[ti][r] = m;
                sA2[bloc * HID + nn] = (_Float16)((m > th) ? 1.f : 0.f);
            }
        }
        __syncthreads();                         // B3: s2 staged (also fences sA1 reads)

        // ---- layer 3: s2 @ W_h2^T via WMMA ----
#pragma unroll
        for (int ti = 0; ti < 4; ++ti) acc[ti] = vzero;
#pragma unroll
        for (int kk = 0; kk < 8; ++kk) {
            Frag32 af;
            af.lo = *(const v4u*)(sA2u + arow * 128 + kk * 16 + laneHi * 4);
            af.hi = *(const v4u*)(sA2u + arow * 128 + kk * 16 + laneHi * 4 + 8);
            v16h a = __builtin_bit_cast(v16h, af);
#pragma unroll
            for (int ti = 0; ti < 4; ++ti) {
                int nn = nBase + ti * 16 + lane16;
                Frag32 bf;
                bf.lo = *(const v4u*)(Wh2U + nn * 128 + kk * 16 + laneHi * 4);
                bf.hi = *(const v4u*)(Wh2U + nn * 128 + kk * 16 + laneHi * 4 + 8);
                v16h b = __builtin_bit_cast(v16h, bf);
                acc[ti] = __builtin_amdgcn_wmma_f32_16x16x32_f16(
                    false, a, false, b, (short)0, acc[ti], false, false);
            }
        }
        // LIF3, spikes s4 -> sA1 (safe: all sA1 readers passed B3)
#pragma unroll
        for (int ti = 0; ti < 4; ++ti) {
            int nn = nBase + ti * 16 + lane16;
            float bt = pL[6 * HID + nn], th = pL[7 * HID + nn], bb = pL[8 * HID + nn];
#pragma unroll
            for (int r = 0; r < 8; ++r) {
                int bloc = rowBase + r + laneHi * 8;
                float cur = acc[ti][r] + bb;
                float m = m4[ti][r];
                float rst = (m > th) ? th : 0.f;
                m = bt * m + cur - rst;
                m4[ti][r] = m;
                sA1[bloc * HID + nn] = (_Float16)((m > th) ? 1.f : 0.f);
            }
        }
        __syncthreads();                         // B4: s4 staged

        // ---- output leaky integrator (OUT=2): all 256 threads split the 256-dot 4-way ----
        {
            float part = 0.f;
            const v4u* sp = (const v4u*)(sA1u + ob * 128 + kc * 16);   // 32 halves, 16B chunks
            const float* wbase = woL + oo * HID + kc * 64;
#pragma unroll
            for (int i = 0; i < 4; ++i) {
                v8h hh = __builtin_bit_cast(v8h, sp[i]);               // 8 spike values
                v4f wa = *(const v4f*)(wbase + i * 8);
                v4f wb = *(const v4f*)(wbase + i * 8 + 4);
                part += (float)hh[0] * wa[0] + (float)hh[1] * wa[1]
                      + (float)hh[2] * wa[2] + (float)hh[3] * wa[3]
                      + (float)hh[4] * wb[0] + (float)hh[5] * wb[1]
                      + (float)hh[6] * wb[2] + (float)hh[7] * wb[3];
            }
            // reduce 4 k-chunks: lanes sharing (b,o) differ only in bits 0-1 of lane id
            part += __shfl_xor(part, 1, 32);
            part += __shfl_xor(part, 2, 32);
            if (kc == 0) {
                m3 = betaO * m3 + part + biasO;                        // reset='none'
                recL[ob * 4 + oo]     = ((m3 - 1.f) > 0.f) ? 1.f : 0.f; // spk_rec
                recL[ob * 4 + 2 + oo] = m3;                             // mem_rec
            }
        }
        __syncthreads();                         // B5: rec channels visible

        if (kc == 0) {
            // out[b, o] accumulates rec[t,b,j] * W_pred[o, 4t+j]
#pragma unroll
            for (int j = 0; j < 4; ++j)
                accP += recL[ob * 4 + j] * wpL[oo * (T_STEPS * 4) + t * 4 + j];
        }
    }

    __syncthreads();
    if (kc == 0)
        out[(size_t)(wgB + ob) * OUTD + oo] = accP + b_pred[oo];
}

extern "C" void kernel_launch(void* const* d_in, const int* in_sizes, int n_in,
                              void* d_out, int out_size, void* d_ws, size_t ws_size,
                              hipStream_t stream) {
    (void)in_sizes; (void)n_in; (void)out_size; (void)d_ws; (void)ws_size;
    const float* x        = (const float*)d_in[0];
    const float* W_in     = (const float*)d_in[1];
    const float* b_in     = (const float*)d_in[2];
    const float* beta_in  = (const float*)d_in[3];
    const float* thr_in   = (const float*)d_in[4];
    const float* W_h      = (const float*)d_in[5];
    const float* b_h      = (const float*)d_in[6];
    const float* beta_h   = (const float*)d_in[7];
    const float* thr_h    = (const float*)d_in[8];
    const float* W_h2     = (const float*)d_in[9];
    const float* b_h2     = (const float*)d_in[10];
    const float* beta_h2  = (const float*)d_in[11];
    const float* thr_h2   = (const float*)d_in[12];
    const float* W_out    = (const float*)d_in[13];
    const float* b_out    = (const float*)d_in[14];
    const float* beta_out = (const float*)d_in[15];
    const float* W_pred   = (const float*)d_in[16];
    const float* b_pred   = (const float*)d_in[17];
    float* out = (float*)d_out;

    // ~305 KB of LDS per workgroup (CDNA5 WGP has 320 KB) — raise the cap.
    (void)hipFuncSetAttribute((const void*)snn_scan_kernel,
                              hipFuncAttributeMaxDynamicSharedMemorySize,
                              (int)SMEM_BYTES);

    snn_scan_kernel<<<NBLK, 256, SMEM_BYTES, stream>>>(
        x, W_in, b_in, beta_in, thr_in,
        W_h, b_h, beta_h, thr_h,
        W_h2, b_h2, beta_h2, thr_h2,
        W_out, b_out, beta_out, W_pred, b_pred, out);
}
